// LSTM_21251498181294
// MI455X (gfx1250) — matrix operand order repaired and checked by reading, not compiled, not run on hardware
//
#include <hip/hip_runtime.h>

// ---------------------------------------------------------------------------
// Bidirectional 2-layer LSTM for MI455X (gfx1250), wave32 + WMMA.
//
// Precision: f16 hi/lo split ("f16x2") with fp32 accumulate:
//   a = hi + lo, acc += Ahi*Bhi + Ahi*Blo + Alo*Bhi  (missing lo*lo ~ 2^-22)
// All splitting is done ONCE per tensor in standalone kernels so the GEMM
// inner loops are pure b128 loads + v_wmma_f32_16x16x32_f16.
//
// Phase 1 (per layer/dir): xg = x @ W_ih^T + b_ih + b_hh  (M=32768, N=2048)
// Phase 2: persistent scan kernel, 32 blocks (16 fwd + 16 bwd), one software
//   grid barrier per step. Each wave owns a 16(batch)x16(hid) tile and
//   accumulates all 4 gate tiles -> gates fused in-register. h state is
//   kept as f16 hi/lo ping-pong buffers (written split, read split).
// ---------------------------------------------------------------------------

#define SEQ   512
#define BATCH 64
#define HID   512
#define G4    2048            // 4*HID
#define OUTW  1024            // 2*HID per (s,b)
#define MROWS (SEQ * BATCH)   // 32768

typedef __attribute__((ext_vector_type(16))) _Float16 v16h;
typedef __attribute__((ext_vector_type(8)))  _Float16 v8h;
typedef __attribute__((ext_vector_type(8)))  float    v8f;

union V16 { v16h v; v8h h[2]; };

__device__ __forceinline__ float sigf(float x) {
    return 1.0f / (1.0f + __expf(-x));
}

// --------------------------- grid-wide barrier -----------------------------
// Assumes all 32 blocks are co-resident (trivial on a many-WGP MI455X).
__device__ __forceinline__ void gridBarrier(unsigned* cnt, unsigned* gen,
                                            unsigned nblk, unsigned& localGen) {
    __syncthreads();
    if (threadIdx.x == 0) {
        unsigned g = localGen;
        __threadfence();
        unsigned arrived = atomicAdd(cnt, 1u);
        if (arrived == nblk - 1u) {
            *cnt = 0u;                 // safe: nobody touches cnt until next gen
            __threadfence();
            atomicAdd(gen, 1u);
        } else {
            while (atomicAdd(gen, 0u) <= g) { __builtin_amdgcn_s_sleep(1); }
        }
        __threadfence();
        localGen = g + 1u;
    }
    __syncthreads();
}

// ------------------- f16 hi/lo split (one pass per tensor) -----------------
__global__ void split_f16_kernel(const float* __restrict__ w,
                                 _Float16* __restrict__ hi,
                                 _Float16* __restrict__ lo, int n) {
    int i = blockIdx.x * blockDim.x + threadIdx.x;
    if (i < n) {
        float v = w[i];
        _Float16 h = (_Float16)v;
        hi[i] = h;
        lo[i] = (_Float16)(v - (float)h);
    }
}

// ------------------ fragment loaders (pre-split f16 data) ------------------
// A fragment (16x32): lane = 16*half + lm; element j -> k:
//   j<8 : k = kb + 16*half + j          (h[0], one 16B load)
//   j>=8: k = kb + 16*half + j          (h[1], one 16B load)
__device__ __forceinline__ void loadA_h16(const _Float16* __restrict__ row,
                                          int kb, int hlf, V16& a) {
    a.h[0] = *(const v8h*)(row + kb + 16 * hlf);
    a.h[1] = *(const v8h*)(row + kb + 16 * hlf + 8);
}
// B fragment (32x16): lane = 16*half + lm; n = lm; element j -> k = kb+16*half+j
// -> 16 contiguous f16 = two 16B loads from row n.
__device__ __forceinline__ void loadB_h16(const _Float16* __restrict__ row,
                                          int kb, int hlf, V16& b) {
    const v8h* p = (const v8h*)(row + kb + 16 * hlf);
    b.h[0] = p[0];
    b.h[1] = p[1];
}

// ------------------------- input-projection GEMM ---------------------------
// xg[m,n] = sum_k A[m,k]*W[n,k] + bi[n] + bh[n];  A,W pre-split f16 hi/lo.
// One wave computes a 16(M) x 64(N) block: inner loop = 20 b128 loads + 12 WMMA.
__global__ __launch_bounds__(256)
void gemm_gates_kernel(const _Float16* __restrict__ Ahi,
                       const _Float16* __restrict__ Alo, int K,
                       const _Float16* __restrict__ Bhi,
                       const _Float16* __restrict__ Blo,
                       const float* __restrict__ bi,
                       const float* __restrict__ bh,
                       float* __restrict__ xg) {
    const int tid  = threadIdx.x;
    const int lane = tid & 31;
    const int lm   = lane & 15;
    const int hlf  = lane >> 4;
    const int gw   = blockIdx.x * 8 + (tid >> 5);   // 65536 waves total
    const int n0   = (gw & 31) * 64;                // 32 N-tiles of 64
    const int m0   = (gw >> 5) * 16;                // 2048 M-tiles of 16

    v8f acc[4] = {};
    const _Float16* arh = Ahi + (size_t)(m0 + lm) * K;
    const _Float16* arl = Alo + (size_t)(m0 + lm) * K;

    for (int kb = 0; kb < K; kb += 32) {
        V16 ahi, alo;
        loadA_h16(arh, kb, hlf, ahi);
        loadA_h16(arl, kb, hlf, alo);
#pragma unroll
        for (int ns = 0; ns < 4; ++ns) {
            const int n = n0 + ns * 16 + lm;
            V16 bhi, blo;
            loadB_h16(Bhi + (size_t)n * K, kb, hlf, bhi);
            loadB_h16(Blo + (size_t)n * K, kb, hlf, blo);
            acc[ns] = __builtin_amdgcn_wmma_f32_16x16x32_f16(
                false, ahi.v, false, bhi.v, (short)0, acc[ns], false, false);
            acc[ns] = __builtin_amdgcn_wmma_f32_16x16x32_f16(
                false, ahi.v, false, blo.v, (short)0, acc[ns], false, false);
            acc[ns] = __builtin_amdgcn_wmma_f32_16x16x32_f16(
                false, alo.v, false, bhi.v, (short)0, acc[ns], false, false);
        }
    }
#pragma unroll
    for (int ns = 0; ns < 4; ++ns) {
        const int n = n0 + ns * 16 + lm;
        const float bias = bi[n] + bh[n];
#pragma unroll
        for (int r = 0; r < 8; ++r) {
            const int m = m0 + r + 8 * hlf;       // C/D layout: M = r + 8*half
            xg[(size_t)m * G4 + n] = acc[ns][r] + bias;
        }
    }
}

// ----------------------------- state init ----------------------------------
// h4 = the 4 contiguous h-split arrays (hi/lo x fwd/bwd, each double-buffered).
__global__ void init_state_kernel(unsigned short* h4, float* cf, float* cb,
                                  unsigned* bar) {
    int i = blockIdx.x * blockDim.x + threadIdx.x;
    if (i < 4 * 2 * BATCH * HID) h4[i] = 0;              // f16 +0.0
    if (i < BATCH * HID) { cf[i] = 0.f; cb[i] = 0.f; }
    if (i == 0)          { bar[0] = 0u; bar[1] = 0u; }
}

// ------------------------- persistent LSTM scan ----------------------------
// 32 blocks x 256 threads; blocks 0..15 fwd, 16..31 bwd. Per dir: 128 waves,
// each owning batch rows [m0,m0+16) x hidden cols [hc0,hc0+16), computing all
// 4 gate tiles. h state stored as f16 hi/lo (ping-pong); c is wave-private.
__global__ __launch_bounds__(256)
void lstm_scan_kernel(const float* __restrict__ xg_f,
                      const float* __restrict__ xg_b,
                      const _Float16* __restrict__ whi_f,
                      const _Float16* __restrict__ wlo_f,
                      const _Float16* __restrict__ whi_b,
                      const _Float16* __restrict__ wlo_b,
                      _Float16* hhi_f, _Float16* hlo_f,
                      _Float16* hhi_b, _Float16* hlo_b,
                      float* cbuf_f, float* cbuf_b,
                      float* __restrict__ out, unsigned* bar) {
    const int tid  = threadIdx.x;
    const int lane = tid & 31;
    const int lm   = lane & 15;
    const int hlf  = lane >> 4;
    const int dir  = blockIdx.x >> 4;                       // 0 fwd, 1 bwd
    const int gw   = ((blockIdx.x & 15) << 3) | (tid >> 5); // 0..127
    const int m0   = (gw & 3) * 16;                         // batch tile
    const int hc0  = (gw >> 2) * 16;                        // hidden-col tile

    const float*    xg  = dir ? xg_b  : xg_f;
    const _Float16* whi = dir ? whi_b : whi_f;
    const _Float16* wlo = dir ? wlo_b : wlo_f;
    _Float16* hhi = dir ? hhi_b : hhi_f;
    _Float16* hlo = dir ? hlo_b : hlo_f;
    float*   cbuf = dir ? cbuf_b : cbuf_f;

    unsigned localGen = 0;

    for (int step = 0; step < SEQ; ++step) {
        const int s = dir ? (SEQ - 1 - step) : step;
        const size_t cur = (size_t)(step & 1) * (BATCH * HID);
        const size_t nxt = (size_t)((step + 1) & 1) * (BATCH * HID);

        v8f acc[4] = {};
        const _Float16* hrh = hhi + cur + (size_t)(m0 + lm) * HID;
        const _Float16* hrl = hlo + cur + (size_t)(m0 + lm) * HID;

        for (int kb = 0; kb < HID; kb += 32) {
            V16 ahi, alo;
            loadA_h16(hrh, kb, hlf, ahi);
            loadA_h16(hrl, kb, hlf, alo);
#pragma unroll
            for (int g = 0; g < 4; ++g) {
                const int n = g * HID + hc0 + lm;           // row of W_hh
                V16 bhi, blo;
                loadB_h16(whi + (size_t)n * HID, kb, hlf, bhi);
                loadB_h16(wlo + (size_t)n * HID, kb, hlf, blo);
                acc[g] = __builtin_amdgcn_wmma_f32_16x16x32_f16(
                    false, ahi.v, false, bhi.v, (short)0, acc[g], false, false);
                acc[g] = __builtin_amdgcn_wmma_f32_16x16x32_f16(
                    false, ahi.v, false, blo.v, (short)0, acc[g], false, false);
                acc[g] = __builtin_amdgcn_wmma_f32_16x16x32_f16(
                    false, alo.v, false, bhi.v, (short)0, acc[g], false, false);
            }
        }

        const float* xgs = xg + (size_t)s * (BATCH * G4);
        const int hc = hc0 + lm;
#pragma unroll
        for (int r = 0; r < 8; ++r) {
            const int b = m0 + r + 8 * hlf;
            const size_t bo = (size_t)b * G4 + hc;
            float gi = acc[0][r] + xgs[bo + 0 * HID];
            float gf = acc[1][r] + xgs[bo + 1 * HID];
            float gg = acc[2][r] + xgs[bo + 2 * HID];
            float go = acc[3][r] + xgs[bo + 3 * HID];
            float c_old = cbuf[(size_t)b * HID + hc];
            float c_new = sigf(gf) * c_old + sigf(gi) * tanhf(gg);
            float h     = sigf(go) * tanhf(c_new);
            cbuf[(size_t)b * HID + hc] = c_new;
            _Float16 hh = (_Float16)h;                       // write h pre-split
            hhi[nxt + (size_t)b * HID + hc] = hh;
            hlo[nxt + (size_t)b * HID + hc] = (_Float16)(h - (float)hh);
            out[(size_t)s * (BATCH * OUTW) + (size_t)b * OUTW + dir * HID + hc] = h;
        }
        gridBarrier(bar, bar + 1, 32, localGen);
    }
}

// ------------------------- final (h, c) assembly ---------------------------
__global__ void finalize_hc_kernel(const float* __restrict__ cf,
                                   const float* __restrict__ cb,
                                   float* __restrict__ out) {
    int i = blockIdx.x * blockDim.x + threadIdx.x;      // [0, 64*1024)
    if (i >= BATCH * OUTW) return;
    int b = i >> 10, j = i & 1023;
    int dir = j >> 9, hc = j & 511;
    const size_t HOFF = (size_t)SEQ * BATCH * OUTW;
    const size_t COFF = HOFF + (size_t)BATCH * OUTW;
    // fwd final h = out row at s=SEQ-1; bwd final h = out row at s=0
    size_t src = (size_t)(dir == 0 ? SEQ - 1 : 0) * (BATCH * OUTW)
               + (size_t)b * OUTW + dir * HID + hc;
    out[HOFF + i] = out[src];
    out[COFF + i] = (dir == 0 ? cf : cb)[(size_t)b * HID + hc];
}

// ---------------------------------------------------------------------------
extern "C" void kernel_launch(void* const* d_in, const int* in_sizes, int n_in,
                              void* d_out, int out_size, void* d_ws, size_t ws_size,
                              hipStream_t stream) {
    (void)out_size; (void)ws_size;
    // ---- locate x (unique size SEQ*BATCH*IN_DIM) and the 16 param tensors --
    const int XN = SEQ * BATCH * 512;                    // 16,777,216
    int xi = 0;
    for (int i = 0; i < n_in; ++i) if (in_sizes[i] == XN) { xi = i; break; }
    const float* x = (const float*)d_in[xi];

    int pidx[16]; int c = 0;
    for (int i = 0; i < n_in && c < 16; ++i) if (i != xi) pidx[c++] = i;

    // Detect pytree flatten convention via layer-1 sizes:
    //   insertion order: [W_ih(2M), W_hh(1M), b, b] per dir, fwd first
    //   sorted-key order: [W_hh(1M), W_ih(2M), b, b] per dir, bwd first
    bool sortedOrder = (in_sizes[pidx[8]] < in_sizes[pidx[9]]);

    struct DirP { const float *wih, *whh, *b0, *b1; };
    DirP P[2][2];                                        // [layer][0=fwd,1=bwd]
    for (int L = 0; L < 2; ++L) {
        for (int d = 0; d < 2; ++d) {
            const int base = L * 8 + d * 4;
            DirP dd;
            dd.wih = (const float*)d_in[pidx[base + (sortedOrder ? 1 : 0)]];
            dd.whh = (const float*)d_in[pidx[base + (sortedOrder ? 0 : 1)]];
            dd.b0  = (const float*)d_in[pidx[base + 2]];   // b order irrelevant,
            dd.b1  = (const float*)d_in[pidx[base + 3]];   // they are summed
            int dirIdx = sortedOrder ? (1 - d) : d;
            P[L][dirIdx] = dd;
        }
    }

    // ---- workspace layout ----
    char* w = (char*)d_ws;
    float* xg_f = (float*)w;                 w += (size_t)MROWS * G4 * 4;  // 256 MB
    float* xg_b = (float*)w;                 w += (size_t)MROWS * G4 * 4;  // 256 MB
    _Float16* ahi = (_Float16*)w;            w += (size_t)MROWS * 1024 * 2; // 64 MB (max K)
    _Float16* alo = (_Float16*)w;            w += (size_t)MROWS * 1024 * 2;
    _Float16* wih_hf = (_Float16*)w;         w += (size_t)G4 * 1024 * 2;    // 4 MB (max K)
    _Float16* wih_lf = (_Float16*)w;         w += (size_t)G4 * 1024 * 2;
    _Float16* wih_hb = (_Float16*)w;         w += (size_t)G4 * 1024 * 2;
    _Float16* wih_lb = (_Float16*)w;         w += (size_t)G4 * 1024 * 2;
    _Float16* whh_hf = (_Float16*)w;         w += (size_t)G4 * HID * 2;     // 2 MB
    _Float16* whh_lf = (_Float16*)w;         w += (size_t)G4 * HID * 2;
    _Float16* whh_hb = (_Float16*)w;         w += (size_t)G4 * HID * 2;
    _Float16* whh_lb = (_Float16*)w;         w += (size_t)G4 * HID * 2;
    _Float16* hst = (_Float16*)w;            w += (size_t)4 * 2 * BATCH * HID * 2; // h splits
    _Float16* hhi_f = hst;                            // each 2*64*512 halves (ping-pong)
    _Float16* hlo_f = hst + 2 * BATCH * HID;
    _Float16* hhi_b = hst + 4 * BATCH * HID;
    _Float16* hlo_b = hst + 6 * BATCH * HID;
    float* cbf = (float*)w;                  w += (size_t)BATCH * HID * 4;
    float* cbb = (float*)w;                  w += (size_t)BATCH * HID * 4;
    unsigned* bar = (unsigned*)w;
    float* out = (float*)d_out;

    const int WHN = G4 * HID;                            // W_hh elements
    for (int L = 0; L < 2; ++L) {
        const float* Ain = (L == 0) ? x : out;           // layer-0 out staged in d_out
        const int K   = (L == 0) ? 512 : 1024;
        const int AN  = MROWS * K;
        const int WIN = G4 * K;
        split_f16_kernel<<<(AN + 255) / 256, 256, 0, stream>>>(Ain, ahi, alo, AN);
        split_f16_kernel<<<(WIN + 255) / 256, 256, 0, stream>>>(P[L][0].wih, wih_hf, wih_lf, WIN);
        split_f16_kernel<<<(WIN + 255) / 256, 256, 0, stream>>>(P[L][1].wih, wih_hb, wih_lb, WIN);
        split_f16_kernel<<<(WHN + 255) / 256, 256, 0, stream>>>(P[L][0].whh, whh_hf, whh_lf, WHN);
        split_f16_kernel<<<(WHN + 255) / 256, 256, 0, stream>>>(P[L][1].whh, whh_hb, whh_lb, WHN);
        gemm_gates_kernel<<<8192, 256, 0, stream>>>(ahi, alo, K, wih_hf, wih_lf,
                                                    P[L][0].b0, P[L][0].b1, xg_f);
        gemm_gates_kernel<<<8192, 256, 0, stream>>>(ahi, alo, K, wih_hb, wih_lb,
                                                    P[L][1].b0, P[L][1].b1, xg_b);
        init_state_kernel<<<1024, 256, 0, stream>>>((unsigned short*)hst, cbf, cbb, bar);
        lstm_scan_kernel<<<32, 256, 0, stream>>>(xg_f, xg_b,
                                                 whh_hf, whh_lf, whh_hb, whh_lb,
                                                 hhi_f, hlo_f, hhi_b, hlo_b,
                                                 cbf, cbb, out, bar);
    }
    finalize_hc_kernel<<<(BATCH * OUTW + 255) / 256, 256, 0, stream>>>(cbf, cbb, out);
}
